// NaiveFourierKANLayer_69080253989643
// MI455X (gfx1250) — matrix operand order repaired and checked by reading, not compiled, not run on hardware
//
#include <hip/hip_runtime.h>
#include <hip/hip_bf16.h>

// ---------------------------------------------------------------------------
// NaiveFourierKANLayer for MI455X (gfx1250, wave32, WMMA)
//   B=16, N=1024, C=4, IN=64, OUT=128, G=16
//
// Stage 0: coeffs (f32) -> f16, pre-blocked into WMMA B-fragment layout.
// Stage 1: A (f32) -> sum over c -> f16, pre-blocked into WMMA A-fragment
//          layout (single streaming pass over the 268 MB tensor = roofline).
// Stage 2: GEMM1 (M=16384,K=2048,N=128): trig features x coeffs -> F (f16,
//          B-fragment blocked).  K-loop split into cos half / sin half so the
//          trig select is branchless.
// Stage 3: GEMM2 per batch (M=1024,K=1024,N=128): A-blk x F-blk + bias -> out.
// ---------------------------------------------------------------------------

typedef __attribute__((ext_vector_type(16))) _Float16 v16h;
typedef __attribute__((ext_vector_type(8)))  _Float16 v8h;
typedef __attribute__((ext_vector_type(8)))  float    v8f;
typedef __attribute__((ext_vector_type(4)))  float    v4f;

union V16U { v16h v; v8h h[2]; };

__device__ __forceinline__ v16h ldg_v16h(const _Float16* p) {
  V16U u;
  u.h[0] = *(const v8h*)(p);
  u.h[1] = *(const v8h*)(p + 8);
  return u.v;
}

#define KB   16
#define KN   1024
#define KC   4
#define KIN  64
#define KOUT 128
#define KG   16

// ---------------------------------------------------------------------------
// Kernel 0: fouriercoeffs[2,OUT,IN,G] f32 -> f16 blocked for WMMA B-fragments.
// addr = (ks*8+ot)*512 + lane*16 + h, o = ot*16+(lane&15), kk=(lane>>4)*16+h,
// k = ks*32+kk = d*1024 + c*16 + g.  (B 32x16 fragment, contiguous k order.)
// ---------------------------------------------------------------------------
__global__ __launch_bounds__(256) void coeff_blk_kernel(
    const float* __restrict__ FC, _Float16* __restrict__ Cblk) {
  int p  = blockIdx.x * 256 + threadIdx.x;       // [0, 64*8*512)
  int h  = p & 15;
  int l  = (p >> 4) & 31;
  int ot = (p >> 9) & 7;
  int ks = p >> 12;
  int o  = ot * 16 + (l & 15);
  int kk = ((l >> 4) << 4) + h;
  int k  = ks * 32 + kk;
  int d  = k >> 10;
  int c  = (k >> 4) & 63;
  int g  = k & 15;
  Cblk[p] = (_Float16)FC[(((d * KOUT + o) * KIN + c) * KG) + g];
}

// ---------------------------------------------------------------------------
// Kernel 1: Ablk fragment = sum_c A[b,j,c,i], f16, in contiguous-k (kk=16*half+h) A 16x32
// layout (M=i, K=j).  addr = ((b*64+ib)*32+jb)*512 + lane*16 + h.
// float4-vectorized: 4 consecutive i per thread -> 4x global_load_b128.
// ---------------------------------------------------------------------------
__global__ __launch_bounds__(256) void reduce_blk_kernel(
    const float* __restrict__ A, _Float16* __restrict__ Ablk) {
  size_t idx = (size_t)blockIdx.x * 256 + threadIdx.x;  // 16*1024*256 threads
  int i0 = (int)(idx & 255) * 4;
  int j  = (int)((idx >> 8) & 1023);
  int b  = (int)(idx >> 18);

  const float* p = A + ((size_t)(b * KN + j) * KC) * KN + i0;
  v4f s = *(const v4f*)(p);
  s += *(const v4f*)(p + KN);
  s += *(const v4f*)(p + 2 * KN);
  s += *(const v4f*)(p + 3 * KN);

  int jb  = j >> 5, kk = j & 31;
  int grp = kk >> 3;                        // 0..3
  int lhi = 16 * (kk >> 4);
  int h   = kk & 15;
  size_t base = ((size_t)(b * 64 + (i0 >> 4)) * 32 + jb) * 512 + h;
#pragma unroll
  for (int t = 0; t < 4; ++t) {
    int m = (i0 + t) & 15;
    Ablk[base + (size_t)(m + lhi) * 16] = (_Float16)s[t];
  }
}

// ---------------------------------------------------------------------------
// GEMM1 K-step (templated on cos/sin so the trig choice is branchless).
// ---------------------------------------------------------------------------
template <bool USE_SIN>
__device__ __forceinline__ void gemm1_step(
    int ks, int tid, int lane, int wave, const float* Vt, _Float16* feat,
    const _Float16* __restrict__ Cblk, v8f& acc) {
  int buf = (ks & 1) * 512;
#pragma unroll
  for (int t = 0; t < 2; ++t) {
    int p  = tid + t * 256;
    int m  = p >> 5, kk = p & 31;
    int c  = ((ks & 31) << 1) + (kk >> 4);
    int g  = kk & 15;
    float x = Vt[m * KIN + c] * (float)(g + 1);
    feat[buf + m * 32 + kk] = (_Float16)(USE_SIN ? __sinf(x) : __cosf(x));
  }
  __syncthreads();

  // A-fragment (16x32 f16) from LDS: two ds_load_b128 per lane
  int m_a = lane & 15, half = lane >> 4;
  V16U ua;
  ua.h[0] = *(const v8h*)&feat[buf + m_a * 32 + half * 16];
  ua.h[1] = *(const v8h*)&feat[buf + m_a * 32 + half * 16 + 8];

  // B-fragment: contiguous 32B per lane from pre-blocked coeffs
  v16h bf = ldg_v16h(Cblk + ((size_t)(ks * 8 + wave) * 512) + lane * 16);

  acc = __builtin_amdgcn_wmma_f32_16x16x32_f16(
      false, ua.v, false, bf, (short)0, acc, false, false);
}

// ---------------------------------------------------------------------------
// Kernel 2 (GEMM1): block = 16 (b,n)-rows x all 128 OUT cols; 8 waves, one
// 16-col OUT tile each.  Features built per 32-wide K step in double-buffered
// LDS (one barrier per step).  Result scattered into B-fragment blocked Fblk.
// ---------------------------------------------------------------------------
__global__ __launch_bounds__(256) void fourier_gemm1_kernel(
    const float* __restrict__ V, const _Float16* __restrict__ Cblk,
    _Float16* __restrict__ Fblk) {
  __shared__ float    Vt[16 * KIN];          // 4 KB
  __shared__ _Float16 feat[2 * 512];         // 2 KB double-buffered

  int tid  = threadIdx.x;
  int lane = tid & 31;
  int wave = tid >> 5;                       // OUT tile index (0..7)
  int nbase = blockIdx.x * 16;               // row base over B*N

  for (int t = tid; t < 16 * KIN; t += 256) {
    int m = t >> 6, c = t & 63;
    Vt[t] = V[(size_t)(nbase + m) * KIN + c];
  }
  __syncthreads();

  v8f acc = {};
  for (int ks = 0; ks < 32; ++ks)            // d=0: cos features
    gemm1_step<false>(ks, tid, lane, wave, Vt, feat, Cblk, acc);
  for (int ks = 32; ks < 64; ++ks)           // d=1: sin features
    gemm1_step<true>(ks, tid, lane, wave, Vt, feat, Cblk, acc);

  // Scatter D tile (C/D layout: VGPR r -> M=r (lanes 0-15) / r+8 (lanes 16-31))
  int o_l = lane & 15;
#pragma unroll
  for (int r = 0; r < 8; ++r) {
    int n  = nbase + r + 8 * (lane >> 4);
    int b  = n >> 10;
    int j  = n & 1023;
    int jb = j >> 5, kk = j & 31;
    int lp = o_l + 16 * (kk >> 4);
    int hp = kk & 15;
    Fblk[(((size_t)(b * 8 + wave) * 32 + jb) * 512) + lp * 16 + hp] =
        (_Float16)acc[r];
  }
}

// ---------------------------------------------------------------------------
// Kernel 3 (GEMM2): out[b,i,o] = sum_j Abar[b,j,i] * F[b,j,o] + bias[o].
// Per block: one batch b, one 16-row i tile; 8 waves cover OUT=128.
// All fragment loads are contiguous 32B/lane (1 KB per wave, fully coalesced).
// ---------------------------------------------------------------------------
__global__ __launch_bounds__(256) void gemm2_kernel(
    const _Float16* __restrict__ Ablk, const _Float16* __restrict__ Fblk,
    const float* __restrict__ bias, float* __restrict__ out) {
  int tid  = threadIdx.x;
  int lane = tid & 31;
  int wave = tid >> 5;                 // OUT tile (0..7)
  int b    = blockIdx.x >> 6;
  int ib   = blockIdx.x & 63;

  const _Float16* ap0 = Ablk + (((size_t)(b * 64 + ib) * 32) * 512) + lane * 16;
  const _Float16* bp0 = Fblk + (((size_t)(b * 8 + wave) * 32) * 512) + lane * 16;

  v8f acc = {};
#pragma unroll 4
  for (int jb = 0; jb < 32; ++jb) {
    v16h a  = ldg_v16h(ap0 + (size_t)jb * 512);
    v16h bb = ldg_v16h(bp0 + (size_t)jb * 512);
    if (jb + 4 < 32) {                  // global_prefetch_b8 hints
      __builtin_prefetch(ap0 + (size_t)(jb + 4) * 512, 0, 0);
      __builtin_prefetch(bp0 + (size_t)(jb + 4) * 512, 0, 0);
    }
    acc = __builtin_amdgcn_wmma_f32_16x16x32_f16(
        false, a, false, bb, (short)0, acc, false, false);
  }

  int o    = wave * 16 + (lane & 15);
  float bv = bias[o];
#pragma unroll
  for (int r = 0; r < 8; ++r) {
    int i = ib * 16 + r + 8 * (lane >> 4);
    out[((size_t)(b * KN + i) * KOUT) + o] = acc[r] + bv;
  }
}

// ---------------------------------------------------------------------------
extern "C" void kernel_launch(void* const* d_in, const int* in_sizes, int n_in,
                              void* d_out, int out_size, void* d_ws,
                              size_t ws_size, hipStream_t stream) {
  (void)in_sizes; (void)n_in; (void)out_size; (void)ws_size;

  const float* V    = (const float*)d_in[0];  // [16,1024,64]
  const float* A    = (const float*)d_in[1];  // [16,1024,4,1024]
  const float* FC   = (const float*)d_in[2];  // [2,128,64,16]
  const float* bias = (const float*)d_in[3];  // [128]
  float*       out  = (float*)d_out;          // [16,1024,128] f32

  char* ws = (char*)d_ws;
  _Float16* Cblk = (_Float16*)ws;                                    // 512 KB
  _Float16* Fblk = (_Float16*)(ws + (512u << 10));                   // 4 MB
  _Float16* Ablk = (_Float16*)(ws + (512u << 10) + (4u << 20));      // 32 MB

  coeff_blk_kernel<<<1024, 256, 0, stream>>>(FC, Cblk);
  reduce_blk_kernel<<<16384, 256, 0, stream>>>(A, Ablk);
  fourier_gemm1_kernel<<<1024, 256, 0, stream>>>(V, Cblk, Fblk);
  gemm2_kernel<<<1024, 256, 0, stream>>>(Ablk, Fblk, bias, out);
}
